// FasterTensorProduct_64063732187600
// MI455X (gfx1250) — matrix operand order repaired and checked by reading, not compiled, not run on hardware
//
#include <hip/hip_runtime.h>

// ---------------------------------------------------------------------------
// FasterTensorProduct on MI455X (gfx1250): bf16 WMMA, f32 accumulate.
// Pair grid: b in [0,4), m (sh row) in [0,128), n (in_ row) in [0,128).
// One workgroup (256 thr = 8 wave32) handles 16 pairs (fixed b,m; 16 n's):
//   1) stage all (pre-scaled, bf16, K-padded) weights into LDS (28 KB)
//   2) build the four feature blocks in LDS as bf16 (f32 math, RNE round)
//   3) 17 output column-tiles, each fully unrolled over K-steps:
//      all ds_load_b128 fragments issued first, then back-to-back
//      v_wmma_f32_16x16x32_bf16 accumulation (C-chained, no hazards)
//   4) scatter 16x16 f32 C tiles straight to the 272-wide output rows
// ---------------------------------------------------------------------------

typedef __attribute__((ext_vector_type(16))) __bf16 v16bf;
typedef __attribute__((ext_vector_type(8)))  float  v8f;

#define IN_DIM   272
#define OUT_DIM  272
#define NPAIR_N  128
// padded bf16 weight layout in d_ws (and LDS):
//   W0e : 96 x 64  @ 0      (src 96x64  * 1/sqrt(96))
//   W1o : 128 x 32 @ 6144   (src 128x32 * 1/sqrt(128))
//   W1e : 96 x 32  @ 10240  (src 80x32  * 1/sqrt(80), rows 80..95 = 0)
//   W0o : 64 x 16  @ 13312  (src 48x16  * 1/sqrt(48), rows 48..63 = 0)
#define WB_TOTAL 14336

__device__ __forceinline__ __bf16 f2bf(float f) {
  unsigned u = __builtin_bit_cast(unsigned, f);
  unsigned r = u + 0x7FFFu + ((u >> 16) & 1u);   // round-to-nearest-even
  unsigned short h = (unsigned short)(r >> 16);
  return __builtin_bit_cast(__bf16, h);
}

// ---------------------------------------------------------------------------
// Phase 0: scale + convert weights to bf16, zero-pad K to multiples of 32.
// ---------------------------------------------------------------------------
__global__ void ftp_prep_weights(const float* __restrict__ w,
                                 __bf16* __restrict__ wb) {
  int idx = blockIdx.x * 256 + threadIdx.x;
  if (idx >= WB_TOTAL) return;
  float v = 0.0f;
  if (idx < 6144) {                       // W0e 96x64, no pad
    v = w[idx] * 0.102062072615966f;      // 1/sqrt(96)
  } else if (idx < 10240) {               // W1o 128x32, no pad
    v = w[idx] * 0.088388347648318f;      // 1/sqrt(128)
  } else if (idx < 13312) {               // W1e 96x32 (src 80x32)
    int rel = idx - 10240;
    if (rel < 80 * 32) v = w[10240 + rel] * 0.111803398874989f;   // 1/sqrt(80)
  } else {                                // W0o 64x16 (src 48x16)
    int rel = idx - 13312;
    if (rel < 48 * 16) v = w[12800 + rel] * 0.144337567297406f;   // 1/sqrt(48)
  }
  wb[idx] = f2bf(v);
}

// ---------------------------------------------------------------------------
// One 16x16 output tile: KSTEPS k-steps of 16x16x32 bf16 WMMA.
// All fragments loaded first (independent ds_load_b128 batch), then the
// WMMA chain on the f32 accumulator.
// ---------------------------------------------------------------------------
template<int KSTEPS, int LDA, int LDB>
__device__ __forceinline__ v8f wmma_tile(const __bf16* __restrict__ A,
                                         const __bf16* __restrict__ Bw,
                                         int ncol0, int lr, int half) {
  v16bf av[KSTEPS];
  v16bf bv[KSTEPS];
  const int arow  = lr * LDA;        // A: lane lr = M row
  const int k0a   = half * 8;        // A: two contiguous 8-elem K runs per lane
  const int krow0 = half * 8;        // B: lane lr = N col, K same order as A
  #pragma unroll
  for (int s = 0; s < KSTEPS; ++s) {
    const int kb = s * 32;
    const __bf16* ap = A + arow + kb + k0a;
    #pragma unroll
    for (int j = 0; j < 8; ++j) {
      av[s][j]     = ap[j];          // K = kb + half*8 + j
      av[s][8 + j] = ap[16 + j];     // K = kb + 16 + half*8 + j
    }
    const __bf16* bp = Bw + (kb + krow0) * LDB + ncol0 + lr;
    #pragma unroll
    for (int j = 0; j < 8; ++j) {
      bv[s][j]     = bp[j * LDB];
      bv[s][8 + j] = bp[(16 + j) * LDB];
    }
  }
  v8f acc = {};
  #pragma unroll
  for (int s = 0; s < KSTEPS; ++s)
    acc = __builtin_amdgcn_wmma_f32_16x16x32_bf16(
        false, av[s], false, bv[s], (short)0, acc, false, false);
  return acc;
}

// C scatter: VGPR i -> M = i + half*8, N = lr; column = colbase + N*stride
__device__ __forceinline__ void scatter_tile(float* __restrict__ out,
                                             size_t rowbase, int colbase,
                                             int stride, int lr, int half,
                                             v8f acc) {
  const int col = colbase + lr * stride;
  #pragma unroll
  for (int i = 0; i < 8; ++i) {
    const int M = i + half * 8;
    out[(rowbase + M) * OUT_DIM + col] = acc[i];
  }
}

// ---------------------------------------------------------------------------
// Phase 1: main fused tensor-product + 4 GEMMs kernel.
// ---------------------------------------------------------------------------
__global__ __launch_bounds__(256, 2)
void ftp_kernel(const float* __restrict__ in_,
                const float* __restrict__ sh,
                const __bf16* __restrict__ wb,
                float* __restrict__ out,
                int nwg_total) {
  // LDS: weights 28672 B + features 26624 B = 55296 B  (<< 320 KB WGP LDS)
  __shared__ __align__(16) __bf16 sW[WB_TOTAL];
  __shared__ __align__(16) __bf16 sF0e[16][96];        // K=96
  __shared__ __align__(16) __bf16 sF1o[3][16][128];    // per comp, K=128
  __shared__ __align__(16) __bf16 sF1e[3][16][96];     // per comp, K=96 (80 live)
  __shared__ __align__(16) __bf16 sF0o[16][64];        // K=64 (48 live)

  const int tid = threadIdx.x;
  const int wg  = blockIdx.x;
  if (wg >= nwg_total) return;
  const int ntile = wg & 7;                // 8 n-tiles of 16
  const int m     = (wg >> 3) & 127;       // sh row
  const int b     = wg >> 10;              // batch

  // ---- stage bf16 weights into LDS (128-bit chunks) ----
  {
    const uint4* src = (const uint4*)wb;          // 28672 B = 1792 x uint4
    uint4* dst = (uint4*)sW;
    for (int i = tid; i < 1792; i += 256) dst[i] = src[i];
  }

  // ---- feature build: 16 threads per pair ----
  const int p   = tid >> 4;                // pair within tile (M row)
  const int sub = tid & 15;
  const int n   = ntile * 16 + p;
  const float* xrow = in_ + (size_t)(b * NPAIR_N + n) * IN_DIM;
  const float* shp  = sh  + (size_t)(b * NPAIR_N + m) * 4;
  const float sh0 = shp[0];
  const float s1[3] = { shp[1], shp[2], shp[3] };
  const float inv_s3 = 0.577350269189626f;
  const float inv_s2 = 0.707106781186548f;
  __builtin_prefetch(xrow, 0, 3);          // global_prefetch_b8: warm WGP$

  // F0e: [x0e*sh0 (64) | dot(x1o,sh1)/sqrt3 (32)]
  for (int k = sub; k < 96; k += 16) {
    float v;
    if (k < 64) v = xrow[k] * sh0;
    else {
      const float* a = xrow + 64 + (k - 64) * 3;
      v = (a[0]*s1[0] + a[1]*s1[1] + a[2]*s1[2]) * inv_s3;
    }
    sF0e[p][k] = f2bf(v);
  }
  // F1o / F1e per vector component c
  const int C1[3] = {1, 2, 0};
  const int C2[3] = {2, 0, 1};
  #pragma unroll
  for (int c = 0; c < 3; ++c) {
    const int c1 = C1[c], c2 = C2[c];
    // F1o: [x0e*sh1c (64) | x1o_c*sh0 (32) | cross(x1e,sh1)_c/sqrt2 (32)]
    for (int k = sub; k < 128; k += 16) {
      float v;
      if (k < 64)      v = xrow[k] * s1[c];
      else if (k < 96) v = xrow[64 + (k - 64) * 3 + c] * sh0;
      else {
        const float* a = xrow + 160 + (k - 96) * 3;   // x1e
        v = (a[c1]*s1[c2] - a[c2]*s1[c1]) * inv_s2;
      }
      sF1o[c][p][k] = f2bf(v);
    }
    // F1e: [cross(x1o,sh1)_c/sqrt2 (32) | x1e_c*sh0 (32) | x0o*sh1c (16) | 0-pad]
    for (int k = sub; k < 96; k += 16) {
      float v;
      if (k < 32) {
        const float* a = xrow + 64 + k * 3;           // x1o
        v = (a[c1]*s1[c2] - a[c2]*s1[c1]) * inv_s2;
      }
      else if (k < 64) v = xrow[160 + (k - 32) * 3 + c] * sh0;
      else if (k < 80) v = xrow[256 + (k - 64)] * s1[c];
      else             v = 0.0f;
      sF1e[c][p][k] = f2bf(v);
    }
  }
  // F0o: [dot(x1e,sh1)/sqrt3 (32) | x0o*sh0 (16) | 0-pad]
  for (int k = sub; k < 64; k += 16) {
    float v;
    if (k < 32) {
      const float* a = xrow + 160 + k * 3;
      v = (a[0]*s1[0] + a[1]*s1[1] + a[2]*s1[2]) * inv_s3;
    }
    else if (k < 48) v = xrow[256 + (k - 32)] * sh0;
    else             v = 0.0f;
    sF0o[p][k] = f2bf(v);
  }

  __syncthreads();

  // ---- WMMA phase: 17 column tiles over 8 waves (scalar-uniform dispatch) --
  const int lane = tid & 31;
  const int wave = __builtin_amdgcn_readfirstlane(tid >> 5);
  const int half = lane >> 4;      // K/M half select per ISA layouts
  const int lr   = lane & 15;
  const size_t rowbase = (size_t)((b * NPAIR_N + m) * NPAIR_N + ntile * 16);

  for (int t = wave; t < 17; t += 8) {
    if (t < 4) {            // y0e: 64 cols @ 0
      const int ncol0 = t * 16;
      v8f acc = wmma_tile<3, 96, 64>(&sF0e[0][0], sW, ncol0, lr, half);
      scatter_tile(out, rowbase, ncol0, 1, lr, half, acc);
    } else if (t < 10) {    // y1o: 32 cols x 3 comps @ 64, (o,c) interleave
      const int comp = (t - 4) >> 1, ncol0 = ((t - 4) & 1) * 16;
      v8f acc = wmma_tile<4, 128, 32>(&sF1o[comp][0][0], sW + 6144, ncol0, lr, half);
      scatter_tile(out, rowbase, 64 + ncol0 * 3 + comp, 3, lr, half, acc);
    } else if (t < 16) {    // y1e: 32 cols x 3 comps @ 160, (o,c) interleave
      const int comp = (t - 10) >> 1, ncol0 = ((t - 10) & 1) * 16;
      v8f acc = wmma_tile<3, 96, 32>(&sF1e[comp][0][0], sW + 10240, ncol0, lr, half);
      scatter_tile(out, rowbase, 160 + ncol0 * 3 + comp, 3, lr, half, acc);
    } else {                // y0o: 16 cols @ 256
      v8f acc = wmma_tile<2, 64, 16>(&sF0o[0][0], sW + 13312, 0, lr, half);
      scatter_tile(out, rowbase, 256, 1, lr, half, acc);
    }
  }
}

// ---------------------------------------------------------------------------
extern "C" void kernel_launch(void* const* d_in, const int* in_sizes, int n_in,
                              void* d_out, int out_size, void* d_ws, size_t ws_size,
                              hipStream_t stream) {
  const float* in_ = (const float*)d_in[0];   // (B,1,128,272) f32
  const float* sh  = (const float*)d_in[1];   // (B,128,1,4)  f32
  const float* w   = (const float*)d_in[2];   // (13568,)     f32
  float* out = (float*)d_out;                 // (B,128,128,272) f32
  __bf16* wb = (__bf16*)d_ws;                 // 28672 B scratch

  const int B = in_sizes[0] / (NPAIR_N * IN_DIM);   // = 4
  const int nwg = B * NPAIR_N * (NPAIR_N / 16);     // = 4096

  ftp_prep_weights<<<(WB_TOTAL + 255) / 256, 256, 0, stream>>>(w, wb);
  ftp_kernel<<<nwg, 256, 0, stream>>>(in_, sh, wb, out, nwg);
}
